// AttentionLoss_SSP_90108413870134
// MI455X (gfx1250) — matrix operand order repaired and checked by reading, not compiled, not run on hardware
//
#include <hip/hip_runtime.h>
#include <hip/hip_bf16.h>
#include <stdint.h>

typedef __attribute__((ext_vector_type(16))) _Float16 v16h;
typedef __attribute__((ext_vector_type(8)))  float    v8f;

#define EPSF 1e-8f
#define LN2F 0.69314718055994530942f

// ---------------------------------------------------------------------------
// Workspace layout (bytes):
//   [0,          63488)  rowmask: uint64[16 * 496]   (per b: level offsets 0,256,384,448,480)
//   [63488,     126976)  colmask: uint64[16 * 496]
//   [126976,    127040)  hasbox : float[16]
//   [127040,    264640)  partials: float[16 * 86 * 25]
//     per (b, level-block) slot of 25 floats: c=0..7 {bce, pt, p} + [24]=t_count
// ---------------------------------------------------------------------------

__device__ __constant__ int kLvlOff[5]  = {0, 256, 384, 448, 480};
__device__ __constant__ int kBlkOff[5]  = {0, 64, 80, 84, 85};   // 86 blocks per image
__device__ __constant__ int kNBlk[5]    = {64, 16, 4, 1, 1};

// ------------------------- prep: rectangles -> bitmasks ---------------------
__global__ __launch_bounds__(64)
void ssp_prep_kernel(const float* __restrict__ bboxs,
                     const int* __restrict__ img_h, const int* __restrict__ img_w,
                     const int* __restrict__ alpha, const int* __restrict__ beta,
                     unsigned long long* __restrict__ rowm,
                     unsigned long long* __restrict__ colm,
                     float* __restrict__ hasbox) {
    const int l = blockIdx.x;           // 0..4
    const int b = blockIdx.y;           // 0..15
    const int n = threadIdx.x;          // 0..63
    const int S = 256 >> l;

    __shared__ int ylo[64], yhi[64], xlo[64], xhi[64];
    __shared__ int anyv;

    const float W = (float)(*img_w);
    const float H = (float)(*img_h);

    const float x1 = bboxs[(b * 64 + n) * 4 + 0];
    const float y1 = bboxs[(b * 64 + n) * 4 + 1];
    const float x2 = bboxs[(b * 64 + n) * 4 + 2];
    const float y2 = bboxs[(b * 64 + n) * 4 + 3];

    const bool valid = (x1 <= W) && (y1 <= H) && (x2 <= W) && (y2 <= H);
    const float area = fabsf((x2 - x1) * (y2 - y1));
    const float side = exp2f((float)(l + *alpha));
    const float min_a = side * side;
    float max_a = side * (float)(*beta);
    max_a *= max_a;
    const bool sel = valid && (area >= min_a) && (area <= max_a);

    const float sx = (float)S / W;
    const float sy = (float)S / H;
    int yi1 = (int)fmaxf(floorf(y1 * sy), 0.0f);
    int yi2 = (int)fminf(ceilf(y2 * sy) + 1.0f, (float)S);
    int xi1 = (int)fmaxf(floorf(x1 * sx), 0.0f);
    int xi2 = (int)fminf(ceilf(x2 * sx) + 1.0f, (float)S);
    if (!sel) { yi1 = 0; yi2 = 0; xi1 = 0; xi2 = 0; }   // row==0 kills the product
    ylo[n] = yi1; yhi[n] = yi2; xlo[n] = xi1; xhi[n] = xi2;

    if (n == 0) anyv = 0;
    __syncthreads();
    if (valid) atomicOr(&anyv, 1);      // idempotent -> deterministic
    __syncthreads();
    if (l == 0 && n == 0) hasbox[b] = anyv ? 1.0f : 0.0f;

    unsigned long long* rB = rowm + b * 496 + kLvlOff[l];
    unsigned long long* cB = colm + b * 496 + kLvlOff[l];
    for (int h = n; h < S; h += 64) {
        unsigned long long m = 0ull;
        #pragma unroll 8
        for (int k = 0; k < 64; ++k)
            if (h >= ylo[k] && h < yhi[k]) m |= (1ull << k);
        rB[h] = m;
    }
    for (int w = n; w < S; w += 64) {
        unsigned long long m = 0ull;
        #pragma unroll 8
        for (int k = 0; k < 64; ++k)
            if (w >= xlo[k] && w < xhi[k]) m |= (1ull << k);
        cB[w] = m;
    }
}

// ------------------- WMMA fragment builders (16-bit layouts) ----------------
// A 16x32 f16: lanes 0-15 -> M=lane, K in {0..7, 16..23}; lanes 16-31 -> K in {8..15, 24..31}
__device__ __forceinline__ v16h build_A(uint32_t bits, int half) {
    v16h m;
    #pragma unroll
    for (int e = 0; e < 16; ++e) {
        const int k = e + ((e < 8) ? 0 : 8) + half * 8;
        m[e] = ((bits >> k) & 1u) ? (_Float16)1.0f : (_Float16)0.0f;
    }
    return m;
}
// B 32x16 f16: lanes 0-15 -> N=lane, K in {0..7, 16..23}; lanes 16-31 -> K in {8..15, 24..31}
__device__ __forceinline__ v16h build_B(uint32_t bits, int half) {
    v16h m;
    #pragma unroll
    for (int e = 0; e < 16; ++e) {
        const int k = e + ((e < 8) ? 0 : 8) + half * 8;
        m[e] = ((bits >> k) & 1u) ? (_Float16)1.0f : (_Float16)0.0f;
    }
    return m;
}

__device__ __forceinline__ float wave_sum32(float v) {
    #pragma unroll
    for (int o = 16; o >= 1; o >>= 1) v += __shfl_xor(v, o, 32);
    return v;
}

// -------------------------- fused mask + loss pass --------------------------
__global__ __launch_bounds__(32)
void ssp_main_kernel(const float* __restrict__ a0, const float* __restrict__ a1,
                     const float* __restrict__ a2, const float* __restrict__ a3,
                     const float* __restrict__ a4,
                     const unsigned long long* __restrict__ rowm,
                     const unsigned long long* __restrict__ colm,
                     float* __restrict__ part) {
    const int b = blockIdx.y;
    const int x = blockIdx.x;           // 0..85 -> (level, block-in-level)
    int l, blkIn;
    if      (x < 64) { l = 0; blkIn = x; }
    else if (x < 80) { l = 1; blkIn = x - 64; }
    else if (x < 84) { l = 2; blkIn = x - 80; }
    else if (x < 85) { l = 3; blkIn = x - 84; }
    else             { l = 4; blkIn = 0; }

    const int S = 256 >> l;
    const int tilesRow = S >> 4;
    const int tpb = (l == 4) ? 1 : 4;   // tiles per block
    const float* attn = (l == 0) ? a0 : (l == 1) ? a1 : (l == 2) ? a2 : (l == 3) ? a3 : a4;

    const int lane = threadIdx.x;
    const int half = lane >> 4;
    const int lm   = lane & 15;

    const unsigned long long* rB = rowm + b * 496 + kLvlOff[l];
    const unsigned long long* cB = colm + b * 496 + kLvlOff[l];

    float bce[8], pt[8], ps[8];
    #pragma unroll
    for (int c = 0; c < 8; ++c) { bce[c] = 0.0f; pt[c] = 0.0f; ps[c] = 0.0f; }
    float tacc = 0.0f;

    const int SS = S * S;

    for (int t = 0; t < tpb; ++t) {               // uniform per wave: EXEC stays all-ones
        const int tile = blkIn * tpb + t;
        const int tY = (tile / tilesRow) * 16;
        const int tX = (tile % tilesRow) * 16;

        const unsigned long long rm = rB[tY + lm];
        const unsigned long long cm = cB[tX + lm];
        const uint32_t rlo = (uint32_t)rm, rhi = (uint32_t)(rm >> 32);
        const uint32_t clo = (uint32_t)cm, chi = (uint32_t)(cm >> 32);

        const v16h A0 = build_A(rlo, half), B0 = build_B(clo, half);
        const v16h A1 = build_A(rhi, half), B1 = build_B(chi, half);

        v8f cc = {};
        cc = __builtin_amdgcn_wmma_f32_16x16x32_f16(false, A0, false, B0,
                                                    (short)0, cc, false, false);
        cc = __builtin_amdgcn_wmma_f32_16x16x32_f16(false, A1, false, B1,
                                                    (short)0, cc, false, false);

        const int w = tX + lm;
        #pragma unroll
        for (int r = 0; r < 8; ++r) {
            const float t01 = (cc[r] > 0.0f) ? 1.0f : 0.0f;   // mask_gt = count > 0
            tacc += t01;
            const int h = tY + r + half * 8;                  // D row = r + 8*half
            const int base0 = (b * 8 * S + h) * S + w;

            // Batch the 8 channel loads first so they issue back-to-back
            // (one loadcnt wait per row instead of one per element).
            float pv[8];
            #pragma unroll
            for (int c = 0; c < 8; ++c)
                pv[c] = attn[base0 + c * SS];

            #pragma unroll
            for (int c = 0; c < 8; ++c) {
                const float p = pv[c];
                const float arg = (t01 != 0.0f) ? p : (1.0f - p);
                // p in (1e-4, 1-1e-4): native v_log_f32 (log2) is safe & accurate here
                const float lg2 = __builtin_amdgcn_logf(arg);
                bce[c] = fmaf(lg2, -LN2F, bce[c]);            // -(t*log p + (1-t)*log(1-p))
                ps[c] += p;
                pt[c] += t01 * p;
            }
        }
    }

    float* dst = part + (b * 86 + kBlkOff[l] + blkIn) * 25;
    #pragma unroll
    for (int c = 0; c < 8; ++c) {
        const float rb = wave_sum32(bce[c]);
        const float rt = wave_sum32(pt[c]);
        const float rp = wave_sum32(ps[c]);
        if (lane == 0) { dst[c * 3 + 0] = rb; dst[c * 3 + 1] = rt; dst[c * 3 + 2] = rp; }
    }
    const float rts = wave_sum32(tacc);
    if (lane == 0) dst[24] = rts;
}

// --------------------- deterministic final reduction ------------------------
__global__ __launch_bounds__(256)
void ssp_final_kernel(const float* __restrict__ part,
                      const float* __restrict__ hasbox,
                      float* __restrict__ out) {
    __shared__ float lossArr[640];     // (b, l, c)
    __shared__ float pim[16];
    const int tid = threadIdx.x;

    for (int idx = tid; idx < 640; idx += blockDim.x) {
        const int b = idx / 40;
        const int rem = idx % 40;
        const int l = rem / 8;
        const int c = rem % 8;
        const int nb = kNBlk[l];
        float sb = 0.0f, st = 0.0f, sp = 0.0f, ts = 0.0f;
        for (int i = 0; i < nb; ++i) {
            const float* p = part + ((b * 86 + kBlkOff[l] + i) * 25);
            sb += p[c * 3 + 0];
            st += p[c * 3 + 1];
            sp += p[c * 3 + 2];
            ts += p[24];
        }
        const float S = (float)(256 >> l);
        const float bceM = sb / (S * S);
        const float dice = 1.0f - (2.0f * st + EPSF) / (sp + ts + EPSF);
        lossArr[idx] = 0.5f * bceM + 0.5f * dice;
    }
    __syncthreads();
    if (tid < 16) {
        float s = 0.0f;
        for (int j = 0; j < 40; ++j) s += lossArr[tid * 40 + j];
        pim[tid] = (s / 40.0f) * hasbox[tid];
    }
    __syncthreads();
    if (tid == 0) {
        float s = 0.0f;
        for (int b = 0; b < 16; ++b) s += pim[b];
        out[0] = s / 16.0f;
    }
}

// ---------------------------------------------------------------------------
extern "C" void kernel_launch(void* const* d_in, const int* in_sizes, int n_in,
                              void* d_out, int out_size, void* d_ws, size_t ws_size,
                              hipStream_t stream) {
    (void)in_sizes; (void)n_in; (void)out_size; (void)ws_size;

    const float* a0    = (const float*)d_in[0];
    const float* a1    = (const float*)d_in[1];
    const float* a2    = (const float*)d_in[2];
    const float* a3    = (const float*)d_in[3];
    const float* a4    = (const float*)d_in[4];
    const float* bbox  = (const float*)d_in[5];
    const int*   img_h = (const int*)d_in[6];
    const int*   img_w = (const int*)d_in[7];
    const int*   alpha = (const int*)d_in[8];
    const int*   beta  = (const int*)d_in[9];

    char* ws = (char*)d_ws;
    unsigned long long* rowm = (unsigned long long*)(ws);
    unsigned long long* colm = (unsigned long long*)(ws + 63488);
    float* hasbox            = (float*)(ws + 126976);
    float* part              = (float*)(ws + 127040);
    float* out               = (float*)d_out;

    ssp_prep_kernel<<<dim3(5, 16), 64, 0, stream>>>(bbox, img_h, img_w, alpha, beta,
                                                    rowm, colm, hasbox);
    ssp_main_kernel<<<dim3(86, 16), 32, 0, stream>>>(a0, a1, a2, a3, a4,
                                                     rowm, colm, part);
    ssp_final_kernel<<<1, 256, 0, stream>>>(part, hasbox, out);
}
